// HiveGNN_15195594293849
// MI455X (gfx1250) — matrix operand order repaired and checked by reading, not compiled, not run on hardware
//
#include <hip/hip_runtime.h>
#include <stdint.h>

// CDNA5 WMMA types (wave32): 16 bf16 = 8 VGPRs, 8 f32 = 8 VGPRs
typedef __attribute__((ext_vector_type(16))) __bf16 v16bf;
typedef __attribute__((ext_vector_type(8)))  float  v8f;

// ---------------------------------------------------------------- utilities
__global__ void fill_kernel(float* __restrict__ p, float v, long long n) {
  long long i = (long long)blockIdx.x * blockDim.x + threadIdx.x;
  long long stride = (long long)gridDim.x * blockDim.x;
  for (; i < n; i += stride) p[i] = v;
}

__global__ void degree_kernel(const int* __restrict__ dst, float* __restrict__ deg,
                              long long E) {
  long long i = (long long)blockIdx.x * blockDim.x + threadIdx.x;
  if (i < E) atomicAdd(&deg[dst[i]], 1.0f);
}

__global__ void rsqrt_kernel(float* __restrict__ d, int n) {
  int i = blockIdx.x * blockDim.x + threadIdx.x;
  if (i < n) d[i] = rsqrtf(d[i]);
}

// ------------------------------------------------------------ WMMA GEMM
// Y[M, ldo](cols colBase..colBase+127) = A[M,128] @ W[128, ldw](col slab)
// Block = 256 threads = 8 waves; each wave: one 16-row strip x 128 cols.
// W panel is converted fp32->bf16 and stored in LDS pre-swizzled into the
// per-lane 32x16 B-fragment order k = 16*(lane>>4) + elem (same as A) so a B-frag
// is one contiguous 32-byte LDS read per lane.
__global__ void __launch_bounds__(256)
gemm_bf16_wmma(const float* __restrict__ A, const float* __restrict__ W,
               float* __restrict__ Y, int M, int ldw, int ldo)
{
  __shared__ __bf16 sW[32768];                 // uses 4 kslabs * 8 ntiles * 32 lanes * 16 elems = 16384
  const int tid = threadIdx.x;
  const int colBase = blockIdx.y * 128;

  // Cooperative load + bf16 convert + swizzle of the 128x128 weight panel.
  // For K-slab kb, K-remainder kr: lane = (kr>=16)*16 + (n&15), elem = kr&15.
  for (int idx = tid; idx < 128 * 128; idx += 256) {
    int K = idx >> 7;                          // 0..127 (rows of W = reduction dim)
    int n = idx & 127;                         // 0..127 (cols within slab) -> coalesced
    float w = W[(size_t)K * ldw + colBase + n];
    int kb = K >> 5, kr = K & 31;
    int lane = ((kr >> 4) << 4) | (n & 15);
    int e = kr & 15;
    int nt = n >> 4;
    sW[(size_t)(((((kb << 3) + nt) << 5) + lane) << 4) + e] = (__bf16)w;
  }
  __syncthreads();

  const int wave  = tid >> 5;
  const int lane  = tid & 31;
  const int strip = blockIdx.x * 8 + wave;     // 16-row strip index
  if (strip >= (M >> 4)) return;               // M is a multiple of 16
  const int row  = (strip << 4) + (lane & 15);
  const int half = lane >> 4;

  v8f zero = {0.f, 0.f, 0.f, 0.f, 0.f, 0.f, 0.f, 0.f};
  v8f acc[8];
#pragma unroll
  for (int nt = 0; nt < 8; ++nt) acc[nt] = zero;

#pragma unroll
  for (int kb = 0; kb < 4; ++kb) {
    // A 16x32 bf16 fragment: lane half selects K {0..15}/{16..31}, contiguous.
    const float* ap = A + (size_t)row * 128 + (kb << 5) + (half << 4);
    v16bf afrag;
#pragma unroll
    for (int i = 0; i < 8; ++i) {
      afrag[i]     = (__bf16)ap[i];
      afrag[i + 8] = (__bf16)ap[i + 8];
    }
#pragma unroll
    for (int nt = 0; nt < 8; ++nt) {
      const v16bf bfrag =
          *(const v16bf*)&sW[(size_t)(((((kb << 3) + nt) << 5) + lane) << 4)];
      acc[nt] = __builtin_amdgcn_wmma_f32_16x16x32_bf16(
          false, afrag, false, bfrag, (short)0, acc[nt], false, false);
    }
  }

  // C/D 16x16 f32 layout: lane holds N = lane&15; VGPR r -> M = r + 8*(lane>>4)
  const int ccol = lane & 15;
  const int crow = (lane >> 4) << 3;
#pragma unroll
  for (int nt = 0; nt < 8; ++nt) {
    float* yp = Y + (size_t)((strip << 4) + crow) * ldo + colBase + (nt << 4) + ccol;
#pragma unroll
    for (int r = 0; r < 8; ++r)
      yp[(size_t)r * ldo] = acc[nt][r];
  }
}

// -------------------------------------------------- edge scatter-aggregate
// One wave per edge (incl. self-loops at idx >= E); 4 cols per lane.
// Both tables are L2-resident (51 MB each, 192 MB L2) -> L2-bound atomics.
__global__ void scatter_kernel(const float* __restrict__ y, float* __restrict__ agg,
                               const int* __restrict__ src, const int* __restrict__ dst,
                               const float* __restrict__ dinv,
                               long long E, long long total)
{
  long long idx = (long long)blockIdx.x * 8 + (threadIdx.x >> 5);
  if (idx >= total) return;
  const int lane = threadIdx.x & 31;
  int s, d;
  if (idx < E) { s = src[idx]; d = dst[idx]; }
  else         { s = d = (int)(idx - E); }     // self loop
  const float w = dinv[s] * dinv[d];
  const float4 v = ((const float4*)(y + (size_t)s * 128))[lane];
  float* ap = agg + (size_t)d * 128 + lane * 4;
  atomicAdd(ap + 0, v.x * w);
  atomicAdd(ap + 1, v.y * w);
  atomicAdd(ap + 2, v.z * w);
  atomicAdd(ap + 3, v.w * w);
}

__global__ void bias_relu_kernel(const float* __restrict__ in, const float* __restrict__ b,
                                 float* __restrict__ out, long long total)
{
  long long i = (long long)blockIdx.x * blockDim.x + threadIdx.x;
  if (i < total) out[i] = fmaxf(in[i] + b[i & 127], 0.0f);
}

// ---------------------------------------------------------- mean pooling
__global__ void pool_kernel(const float* __restrict__ agg, const float* __restrict__ b2,
                            const int* __restrict__ batch, float* __restrict__ pooled,
                            float* __restrict__ counts, int N)
{
  int node = blockIdx.x * 8 + (threadIdx.x >> 5);
  if (node >= N) return;
  const int lane = threadIdx.x & 31;
  const int g = batch[node];
  const float4 v  = ((const float4*)(agg + (size_t)node * 128))[lane];
  const float4 bb = ((const float4*)b2)[lane];
  float* pp = pooled + (size_t)g * 128 + lane * 4;
  atomicAdd(pp + 0, fmaxf(v.x + bb.x, 0.0f));
  atomicAdd(pp + 1, fmaxf(v.y + bb.y, 0.0f));
  atomicAdd(pp + 2, fmaxf(v.z + bb.z, 0.0f));
  atomicAdd(pp + 3, fmaxf(v.w + bb.w, 0.0f));
  if (lane == 0) atomicAdd(&counts[g], 1.0f);
}

__global__ void pool_div_kernel(float* __restrict__ pooled, const float* __restrict__ counts,
                                int total)
{
  int i = blockIdx.x * blockDim.x + threadIdx.x;
  if (i < total) pooled[i] /= fmaxf(counts[i >> 7], 1.0f);
}

// ----------------------------------------------------------------- heads
__global__ void value_kernel(const float* __restrict__ pooled, const float* __restrict__ Wv,
                             const float* __restrict__ bv, float* __restrict__ out, int Bn)
{
  int b = blockIdx.x * blockDim.x + threadIdx.x;
  if (b >= Bn) return;
  float acc = bv[0];
  const float* row = pooled + (size_t)b * 128;
#pragma unroll 8
  for (int k = 0; k < 128; ++k) acc += row[k] * Wv[k];
  out[b] = tanhf(acc);
}

__global__ void softmax_kernel(const float* __restrict__ logits, const float* __restrict__ bp,
                               float* __restrict__ out, int A_)
{
  __shared__ float red[256];
  const int b = blockIdx.x, tid = threadIdx.x;
  const float* row = logits + (size_t)b * A_;
  float m = -3.4e38f;
  for (int i = tid; i < A_; i += 256) m = fmaxf(m, row[i] + bp[i]);
  red[tid] = m; __syncthreads();
  for (int s = 128; s > 0; s >>= 1) {
    if (tid < s) red[tid] = fmaxf(red[tid], red[tid + s]);
    __syncthreads();
  }
  m = red[0]; __syncthreads();
  float sum = 0.f;
  for (int i = tid; i < A_; i += 256) sum += __expf(row[i] + bp[i] - m);
  red[tid] = sum; __syncthreads();
  for (int s = 128; s > 0; s >>= 1) {
    if (tid < s) red[tid] += red[tid + s];
    __syncthreads();
  }
  const float inv = 1.0f / red[0];
  for (int i = tid; i < A_; i += 256)
    out[(size_t)b * A_ + i] = __expf(row[i] + bp[i] - m) * inv;
}

// ------------------------------------------------------------------ launch
extern "C" void kernel_launch(void* const* d_in, const int* in_sizes, int n_in,
                              void* d_out, int out_size, void* d_ws, size_t ws_size,
                              hipStream_t stream)
{
  const float* x    = (const float*)d_in[0];
  const int*   eidx = (const int*)d_in[1];
  const int*   batch= (const int*)d_in[2];
  const float* W1   = (const float*)d_in[3];
  const float* b1   = (const float*)d_in[4];
  const float* W2   = (const float*)d_in[5];
  const float* b2   = (const float*)d_in[6];
  const float* Wv   = (const float*)d_in[7];
  const float* bv   = (const float*)d_in[8];
  const float* Wp   = (const float*)d_in[9];
  const float* bp   = (const float*)d_in[10];

  const long long N  = in_sizes[2];      // nodes   (100000, multiple of 16)
  const long long E  = in_sizes[1] / 2;  // edges
  const int A_ = in_sizes[10];           // actions (1024)
  const int Bn = out_size / (A_ + 1);    // graphs  (256)
  const long long NH = N * 128;

  const int* src = eidx;
  const int* dst = eidx + E;

  // workspace layout (fp32): bufA | bufB | dinv | pooled | counts | logits
  float* bufA   = (float*)d_ws;
  float* bufB   = bufA + (size_t)NH;
  float* dinv   = bufB + (size_t)NH;
  float* pooled = dinv + (size_t)N;
  float* counts = pooled + (size_t)Bn * 128;
  float* logits = counts + (size_t)Bn;

  float* out_v = (float*)d_out;
  float* out_p = out_v + Bn;

  dim3 blk(256);
  const long long total = E + N;

  // degrees (self-loop => init 1.0) -> dinv = rsqrt(deg)
  fill_kernel<<<1024, blk, 0, stream>>>(dinv, 1.0f, N);
  degree_kernel<<<(unsigned)((E + 255) / 256), blk, 0, stream>>>(dst, dinv, E);
  rsqrt_kernel<<<(unsigned)((N + 255) / 256), blk, 0, stream>>>(dinv, (int)N);

  // layer 1: y1 = x @ W1  -> bufA
  {
    dim3 g((unsigned)(((N >> 4) + 7) / 8), 1);
    gemm_bf16_wmma<<<g, blk, 0, stream>>>(x, W1, bufA, (int)N, 128, 128);
  }
  fill_kernel<<<2048, blk, 0, stream>>>(bufB, 0.0f, NH);
  scatter_kernel<<<(unsigned)((total + 7) / 8), blk, 0, stream>>>(bufA, bufB, src, dst,
                                                                  dinv, E, total);
  bias_relu_kernel<<<(unsigned)((NH + 255) / 256), blk, 0, stream>>>(bufB, b1, bufA, NH);

  // layer 2: y2 = h1 @ W2 -> bufB
  {
    dim3 g((unsigned)(((N >> 4) + 7) / 8), 1);
    gemm_bf16_wmma<<<g, blk, 0, stream>>>(bufA, W2, bufB, (int)N, 128, 128);
  }
  fill_kernel<<<2048, blk, 0, stream>>>(bufA, 0.0f, NH);
  scatter_kernel<<<(unsigned)((total + 7) / 8), blk, 0, stream>>>(bufB, bufA, src, dst,
                                                                  dinv, E, total);

  // mean pool (bias+relu fused)
  fill_kernel<<<64, blk, 0, stream>>>(pooled, 0.0f, (long long)Bn * 128 + Bn);
  pool_kernel<<<(unsigned)((N + 7) / 8), blk, 0, stream>>>(bufA, b2, batch, pooled,
                                                           counts, (int)N);
  pool_div_kernel<<<(unsigned)((Bn * 128 + 255) / 256), blk, 0, stream>>>(pooled, counts,
                                                                          Bn * 128);

  // heads
  value_kernel<<<(unsigned)((Bn + 255) / 256), blk, 0, stream>>>(pooled, Wv, bv, out_v, Bn);
  {
    dim3 g((unsigned)(((Bn >> 4) + 7) / 8), (unsigned)(A_ / 128));
    gemm_bf16_wmma<<<g, blk, 0, stream>>>(pooled, Wp, logits, Bn, A_, A_);
  }
  softmax_kernel<<<Bn, blk, 0, stream>>>(logits, bp, out_p, A_);
}
